// SimplifiedTransducerSearcher_59416577573506
// MI455X (gfx1250) — compile-verified, matrix operand order checked by reading, not run on hardware
//
#include <hip/hip_runtime.h>
#include <hip/hip_bf16.h>

typedef __attribute__((ext_vector_type(16))) __bf16 v16bf;
typedef __attribute__((ext_vector_type(8)))  float  v8f;
typedef unsigned short ushort_t;
typedef unsigned int   uint_t;

// ---------------------------------------------------------------- helpers

__global__ void f32_to_bf16_kernel(const float* __restrict__ x,
                                   ushort_t* __restrict__ y, long long n) {
    long long i = (long long)blockIdx.x * blockDim.x + threadIdx.x;
    if (i < n) {
        uint_t u = __float_as_uint(x[i]);
        uint_t r = u + 0x7FFFu + ((u >> 16) & 1u);   // round-to-nearest-even
        y[i] = (ushort_t)(r >> 16);
    }
}

__global__ void init_state_kernel(const int* __restrict__ hyps0,
                                  int* __restrict__ tok,
                                  float* __restrict__ sp, int n) {
    int i = blockIdx.x * blockDim.x + threadIdx.x;
    if (i < n) { tok[i] = hyps0[i]; sp[i] = 0.0f; }
}

__global__ void finalize_scores_kernel(const float* __restrict__ sp,
                                       float* __restrict__ out_scores, int B) {
    int b = blockIdx.x * blockDim.x + threadIdx.x;
    if (b < B) out_scores[b] = sp[2 * b] + sp[2 * b + 1];
}

// ---------------------------------------------------------------- WMMA GEMM
// C[Mrows x N] (f32) = A[Mrows x K] (bf16) * B[K x N] (bf16)
// A row r lives at: Abase + (r / rows_inner)*outer_stride + (r % rows_inner)*inner_stride
// (lets one kernel serve both Temb@Wt and the strided audio-chunk GEMM)

__device__ __forceinline__ v16bf load_frag16(const ushort_t (*S)[40],
                                             int rowBase, int lrow, int half) {
    union { uint_t u[8]; v16bf v; } f;
    const ushort_t* row = S[rowBase + lrow];
#pragma unroll
    for (int p = 0; p < 8; ++p) {
        // ISA 16-bit operand layout: VGPR p holds packed K pair
        //  p<4 : K = 2p   + 8*half   (lanes 0-15 -> half=0 ; 16-31 -> half=1)
        //  p>=4: K = 16+2(p-4) + 8*half
        int k = ((p < 4) ? (2 * p) : (16 + 2 * (p - 4))) + 8 * half;
        f.u[p] = *(const uint_t*)&row[k];          // even k -> 4B aligned
    }
    return f.v;
}

__global__ void __launch_bounds__(256)
gemm_bf16_wmma_kernel(const ushort_t* __restrict__ A,
                      const ushort_t* __restrict__ Bm,
                      float* __restrict__ Cm,
                      int Mrows, int N, int K,
                      int rows_inner, long long inner_stride, long long outer_stride) {
    __shared__ alignas(16) ushort_t As[128][40];   // row stride 80B (16B multiple)
    __shared__ alignas(16) ushort_t Bs[64][40];    // stored transposed: [n][k]

    const int tid   = threadIdx.x;
    const int lane  = tid & 31;
    const int wave  = tid >> 5;
    const int lrow  = lane & 15;
    const int half  = lane >> 4;
    const int wm    = wave & 3;        // 4 waves along M  -> 128 rows
    const int wn    = wave >> 2;       // 2 waves along N  -> 64 cols
    const int blockM = blockIdx.x;
    const int blockN = blockIdx.y;

    v8f acc[2][2] = {};

    for (int k0 = 0; k0 < K; k0 += 32) {
        // ---- stage A tile: 128 rows x 32 K via CDNA5 async copy (ASYNCcnt path)
#pragma unroll
        for (int pass = 0; pass < 2; ++pass) {
            int rl = (tid >> 2) + pass * 64;       // 0..127
            int cl = (tid & 3) * 8;                // 0,8,16,24
            int r  = blockM * 128 + rl;
            unsigned lds_a = (unsigned)(unsigned long long)&As[rl][cl]; // low 32b = LDS offset
            if (r < Mrows) {
                const ushort_t* p = A
                    + (long long)(r / rows_inner) * outer_stride
                    + (long long)(r % rows_inner) * inner_stride
                    + k0 + cl;
                asm volatile("global_load_async_to_lds_b128 %0, %1, off"
                             :: "v"(lds_a), "v"(p) : "memory");
            } else {
                *(uint4*)&As[rl][cl] = make_uint4(0u, 0u, 0u, 0u);
            }
        }
        // ---- stage B tile transposed: Bs[n][k], 32 K x 64 N
        {
            int kk = tid >> 3;                     // 0..31
            int nl = (tid & 7) * 8;                // 0..56
            const ushort_t* p = Bm + (long long)(k0 + kk) * N + blockN * 64 + nl;
            uint4 val = *(const uint4*)p;
            if (k0 + 32 < K) __builtin_prefetch(p + (long long)32 * N, 0, 3);
            const ushort_t* t8 = (const ushort_t*)&val;
#pragma unroll
            for (int i = 0; i < 8; ++i) Bs[nl + i][kk] = t8[i];
        }
        // async A-tile copies must land before any wave reads the LDS tile
        asm volatile("s_wait_asynccnt 0" ::: "memory");
        __syncthreads();

        v16bf a0 = load_frag16(As, wm * 32 +  0, lrow, half);
        v16bf a1 = load_frag16(As, wm * 32 + 16, lrow, half);
        v16bf b0 = load_frag16(Bs, wn * 32 +  0, lrow, half);
        v16bf b1 = load_frag16(Bs, wn * 32 + 16, lrow, half);

        acc[0][0] = __builtin_amdgcn_wmma_f32_16x16x32_bf16(false, a0, false, b0, (short)0, acc[0][0], false, false);
        acc[0][1] = __builtin_amdgcn_wmma_f32_16x16x32_bf16(false, a0, false, b1, (short)0, acc[0][1], false, false);
        acc[1][0] = __builtin_amdgcn_wmma_f32_16x16x32_bf16(false, a1, false, b0, (short)0, acc[1][0], false, false);
        acc[1][1] = __builtin_amdgcn_wmma_f32_16x16x32_bf16(false, a1, false, b1, (short)0, acc[1][1], false, false);

        __syncthreads();
    }

    // ---- write back: VGPR r -> (M = r + 8*half, N = lrow) within each 16x16 tile
#pragma unroll
    for (int i = 0; i < 2; ++i) {
#pragma unroll
        for (int j = 0; j < 2; ++j) {
#pragma unroll
            for (int r = 0; r < 8; ++r) {
                int grow = blockM * 128 + wm * 32 + i * 16 + 8 * half + r;
                int gcol = blockN * 64  + wn * 32 + j * 16 + lrow;
                if (grow < Mrows)
                    Cm[(long long)grow * N + gcol] = acc[i][j][r];
            }
        }
    }
}

// ---------------------------------------------------------------- greedy scan
// One block per (b,c) channel; channels are independent across the whole scan.
// Per step: reduce over V of  x[v] = Achunk[b,ti,c,v] + M[tok][v]
//   vals = max(logp) = -log(sum exp(x - max x)) ;  ids = argmax x

__global__ void __launch_bounds__(256)
scan_chunk_kernel(const float* __restrict__ Achunk,
                  const float* __restrict__ Mmat,
                  const float* __restrict__ lens,
                  int* __restrict__ tok_state,
                  float* __restrict__ score_part,
                  float* __restrict__ hyps_out,
                  int t0, int chunkT, int B, int T, int C, int V) {
    const int bc  = blockIdx.x;
    const int b   = bc / C;
    const int c   = bc % C;
    const int tid = threadIdx.x;

    __shared__ float red_m[256];
    __shared__ float red_s[256];
    __shared__ int   red_i[256];
    __shared__ int   s_tok;

    if (tid == 0) s_tok = tok_state[bc];
    __syncthreads();

    const int abs_len = (int)(T * lens[b]);
    const int V4 = V >> 2;                         // V divisible by 4
    float score = 0.0f;

    for (int ti = 0; ti < chunkT; ++ti) {
        const int t = t0 + ti;
        const bool alive = (t < abs_len);          // uniform across block
        if (!alive) {
            if (tid == 0) hyps_out[((long long)b * T + t) * C + c] = 0.0f;
            continue;
        }
        const float4* a4 = (const float4*)(Achunk + (((long long)(b * chunkT + ti)) * C + c) * V);
        const float4* m4 = (const float4*)(Mmat + (long long)s_tok * V);

        float m = -INFINITY, s = 0.0f;
        int   idx = 0;
        for (int v4 = tid; v4 < V4; v4 += 256) {
            float4 xa = a4[v4];
            float4 xm = m4[v4];
            float xs[4] = { xa.x + xm.x, xa.y + xm.y, xa.z + xm.z, xa.w + xm.w };
#pragma unroll
            for (int j = 0; j < 4; ++j) {
                float x = xs[j];
                if (x > m) { s = s * expf(m - x) + 1.0f; m = x; idx = 4 * v4 + j; }
                else       { s += expf(x - m); }
            }
        }
        red_m[tid] = m; red_s[tid] = s; red_i[tid] = idx;
        __syncthreads();
#pragma unroll
        for (int stride = 128; stride > 0; stride >>= 1) {
            if (tid < stride) {
                float m2 = red_m[tid + stride], s2 = red_s[tid + stride];
                int   i2 = red_i[tid + stride];
                float m1 = red_m[tid], s1 = red_s[tid];
                int   i1 = red_i[tid];
                if (m2 > m1 || (m2 == m1 && i2 < i1)) {
                    red_s[tid] = s1 * expf(m1 - m2) + s2;
                    red_m[tid] = m2; red_i[tid] = i2;
                } else {
                    red_s[tid] = s1 + s2 * expf(m2 - m1);
                }
            }
            __syncthreads();
        }
        if (tid == 0) {
            int   ids = red_i[0];
            float val = -logf(red_s[0]);           // max(logp) = max - LSE
            score += val;
            s_tok = ids;
            hyps_out[((long long)b * T + t) * C + c] = (float)ids;
        }
        __syncthreads();                            // publish s_tok for next step
    }

    if (tid == 0) {
        tok_state[bc] = s_tok;
        score_part[bc] += score;
    }
}

// ---------------------------------------------------------------- launch

extern "C" void kernel_launch(void* const* d_in, const int* in_sizes, int n_in,
                              void* d_out, int out_size, void* d_ws, size_t ws_size,
                              hipStream_t stream) {
    const int B = 64, T = 512, D = 1024, C = 2, V = 8000;
    const int chunkT = 32;

    const float* audio = (const float*)d_in[0];   // [B,T,D]
    const float* lens  = (const float*)d_in[1];   // [B]
    const int*   hyps0 = (const int*)d_in[2];     // [B,C]
    const float* Wa    = (const float*)d_in[3];   // [D,C,V]
    const float* Temb  = (const float*)d_in[4];   // [V,D]
    const float* Wt    = (const float*)d_in[5];   // [D,V]

    float* out        = (float*)d_out;
    float* out_hyps   = out;                        // B*T*C
    float* out_scores = out + (size_t)B * T * C;    // B

    char*  ws  = (char*)d_ws;
    size_t off = 0;
    auto alloc = [&](size_t bytes) -> void* {
        void* p = ws + off;
        off = (off + bytes + 255) & ~(size_t)255;
        return p;
    };
    ushort_t* audio_bf = (ushort_t*)alloc((size_t)B * T * D * 2);
    ushort_t* wa_bf    = (ushort_t*)alloc((size_t)D * C * V * 2);
    ushort_t* temb_bf  = (ushort_t*)alloc((size_t)V * D * 2);
    ushort_t* wt_bf    = (ushort_t*)alloc((size_t)D * V * 2);
    float*    Mmat     = (float*)alloc((size_t)V * V * 4);
    float*    Achunk   = (float*)alloc((size_t)B * chunkT * C * V * 4);
    int*      tok      = (int*)alloc((size_t)B * C * 4);
    float*    sp       = (float*)alloc((size_t)B * C * 4);
    (void)ws_size; (void)in_sizes; (void)n_in; (void)out_size;

    // 1) fp32 -> bf16 staging
    {
        long long n;
        n = (long long)B * T * D;
        f32_to_bf16_kernel<<<(unsigned)((n + 255) / 256), 256, 0, stream>>>(audio, audio_bf, n);
        n = (long long)D * C * V;
        f32_to_bf16_kernel<<<(unsigned)((n + 255) / 256), 256, 0, stream>>>(Wa, wa_bf, n);
        n = (long long)V * D;
        f32_to_bf16_kernel<<<(unsigned)((n + 255) / 256), 256, 0, stream>>>(Temb, temb_bf, n);
        n = (long long)D * V;
        f32_to_bf16_kernel<<<(unsigned)((n + 255) / 256), 256, 0, stream>>>(Wt, wt_bf, n);
    }
    init_state_kernel<<<1, 128, 0, stream>>>(hyps0, tok, sp, B * C);

    // 2) M = Temb @ Wt   (V x V, K = D)
    {
        dim3 g((V + 127) / 128, V / 64);
        gemm_bf16_wmma_kernel<<<g, 256, 0, stream>>>(
            temb_bf, wt_bf, Mmat, V, V, D, V, (long long)D, 0LL);
    }

    // 3) per T-chunk: audio logits GEMM, then sequential greedy scan
    for (int t0 = 0; t0 < T; t0 += chunkT) {
        dim3 g((B * chunkT) / 128, (C * V) / 64);
        gemm_bf16_wmma_kernel<<<g, 256, 0, stream>>>(
            audio_bf + (long long)t0 * D, wa_bf, Achunk,
            B * chunkT, C * V, D,
            chunkT, (long long)D, (long long)T * D);
        scan_chunk_kernel<<<B * C, 256, 0, stream>>>(
            Achunk, Mmat, lens, tok, sp, out_hyps, t0, chunkT, B, T, C, V);
    }

    // 4) scores[b] = sum over channels
    finalize_scores_kernel<<<1, 64, 0, stream>>>(sp, out_scores, B);
}
